// DualModeAttention_87634512707678
// MI455X (gfx1250) — matrix operand order repaired and checked by reading, not compiled, not run on hardware
//
#include <hip/hip_runtime.h>
#include <hip/hip_bf16.h>
#include <cstdint>

// ---------------------------------------------------------------------------
// Problem constants
// ---------------------------------------------------------------------------
#define B_   8
#define T_   1024
#define DIM_ 2048
#define H_   16
#define NKV_ 4
#define HD_  128
#define G_   (H_ / NKV_)          // 4
#define MT_  (B_ * T_)            // 8192 rows
#define SCALE_ 0.08838834764831845f   // HD^-0.5
#define EPS_   1e-6f
#define LN_BASE_ 9.210340371976184f   // ln(10000)

typedef __bf16 bf16_t;
typedef __attribute__((ext_vector_type(16))) __bf16 bf16x16;
typedef __attribute__((ext_vector_type(8)))  float  f32x8;
typedef __attribute__((ext_vector_type(4)))  unsigned int u32x4;
typedef __attribute__((ext_vector_type(8)))  int i32x8;
typedef __attribute__((ext_vector_type(4)))  int i32x4;

union ABFrag {
    bf16x16        v;
    unsigned int   u32[8];
    unsigned short u16[16];
};

// ---------------------------------------------------------------------------
// Tensor Data Mover: global -> LDS tile DMA (CDNA5 TDM, TENSORcnt-tracked)
// D# packed per ISA 8.3/8.4:
//  group0: count=1 | lds_addr | global_addr[56:0] | type=2
//  group1: data_size=4B, tensor_dim0/1 (large, tiles always in-bounds),
//          tile_dim0 (dwords), tile_dim1 (rows), tensor_dim0_stride (dwords)
// lds_addr = low 32 bits of flat pointer (local->flat cast keeps LDS offset).
// ---------------------------------------------------------------------------
__device__ __forceinline__ unsigned lds_addr_of(const void* p) {
    return (unsigned)(uintptr_t)p;
}

__device__ __forceinline__ void tdm_load_2d(unsigned lds_addr, const void* gptr,
                                            unsigned tile0_dw, unsigned tile1,
                                            unsigned long long stride_dw) {
    const unsigned long long ga = (unsigned long long)(uintptr_t)gptr;
    u32x4 g0;
    g0[0] = 1u;                                            // count=1 (valid)
    g0[1] = lds_addr;                                      // lds_addr
    g0[2] = (unsigned)(ga & 0xFFFFFFFFull);                // global_addr lo
    g0[3] = (unsigned)((ga >> 32) & 0x01FFFFFFull) | (2u << 30); // hi | type=2

    const unsigned td0 = 1u << 20;                         // tensor_dim0 (big)
    const unsigned td1 = 1u << 20;                         // tensor_dim1 (big)
    i32x8 g1;
    g1[0] = (int)(2u << 16);                               // wg_mask=0, data_size=4B
    g1[1] = (int)((td0 & 0xFFFFu) << 16);                  // atomic_addr=0 | td0 lo16
    g1[2] = (int)((td0 >> 16) | ((td1 & 0xFFFFu) << 16));  // td0 hi16 | td1 lo16
    g1[3] = (int)((td1 >> 16) | (tile0_dw << 16));         // td1 hi16 | tile_dim0
    g1[4] = (int)(tile1 & 0xFFFFu);                        // tile_dim1 | tile_dim2=0
    g1[5] = (int)(unsigned)(stride_dw & 0xFFFFFFFFull);    // dim0 stride lo32
    g1[6] = (int)(unsigned)((stride_dw >> 32) & 0xFFFFull);// dim0 stride hi16
    g1[7] = 0;

    const i32x4 z4 = {0, 0, 0, 0};
#if defined(__clang_major__) && (__clang_major__ >= 23)
    const i32x8 z8 = {0, 0, 0, 0, 0, 0, 0, 0};
    __builtin_amdgcn_tensor_load_to_lds(g0, g1, z4, z4, z8, 0);
#else
    __builtin_amdgcn_tensor_load_to_lds(g0, g1, z4, z4, 0);
#endif
}

// ---------------------------------------------------------------------------
// fp32 -> bf16 conversion
// ---------------------------------------------------------------------------
__global__ void f32_to_bf16_kernel(const float* __restrict__ src,
                                   bf16_t* __restrict__ dst, long n) {
    long i = (long)blockIdx.x * blockDim.x + threadIdx.x;
    if (i < n) dst[i] = (bf16_t)src[i];
}

// ---------------------------------------------------------------------------
// C[M,N] (f32) = A[M,K] (bf16) * W[N,K]^T (bf16)
// 64x64 block tile, 128 threads (4 waves), K-step 32, WMMA bf16.
// Tiles arrive via TDM (double-buffered); wave 0 issues DMA + waits TENSORcnt.
// ---------------------------------------------------------------------------
__global__ __launch_bounds__(128)
void gemm_bf16_wmma(const bf16_t* __restrict__ A,
                    const bf16_t* __restrict__ W,
                    float* __restrict__ C, int M, int N, int K) {
    __shared__ bf16_t As[2][64 * 32];
    __shared__ bf16_t Bs[2][64 * 32];
    const int tid  = threadIdx.x;
    const int lane = tid & 31;
    const int wave = tid >> 5;
    const int l15  = lane & 15;
    const int grp  = lane >> 4;
    const int m0   = blockIdx.y * 64;
    const int n0   = blockIdx.x * 64;
    const unsigned long long strideDw = (unsigned long long)(K >> 1);

    f32x8 acc[4];
    #pragma unroll
    for (int i = 0; i < 4; ++i)
        #pragma unroll
        for (int j = 0; j < 8; ++j) acc[i][j] = 0.0f;

    const int nk = K / 32;
    if (wave == 0) {
        tdm_load_2d(lds_addr_of(&As[0][0]), A + (size_t)m0 * K, 16, 64, strideDw);
        tdm_load_2d(lds_addr_of(&Bs[0][0]), W + (size_t)n0 * K, 16, 64, strideDw);
    }

    for (int kt = 0; kt < nk; ++kt) {
        const int cur = kt & 1;
        if (wave == 0) __builtin_amdgcn_s_wait_tensorcnt(0);
        __syncthreads();                       // tile kt is visible to all waves
        if (wave == 0 && kt + 1 < nk) {        // prefetch next tile via TDM
            tdm_load_2d(lds_addr_of(&As[cur ^ 1][0]),
                        A + (size_t)m0 * K + (kt + 1) * 32, 16, 64, strideDw);
            tdm_load_2d(lds_addr_of(&Bs[cur ^ 1][0]),
                        W + (size_t)n0 * K + (kt + 1) * 32, 16, 64, strideDw);
        }

        ABFrag bf;                             // B frag: this wave's 16 cols
        {
            int n = wave * 16 + l15;
            #pragma unroll
            for (int p = 0; p < 8; ++p) {
                int kk = grp * 16 + p * 2;
                bf.u32[p] = *(const unsigned int*)(&Bs[cur][n * 32 + kk]);
            }
        }
        #pragma unroll
        for (int strip = 0; strip < 4; ++strip) {
            ABFrag af;
            int m = strip * 16 + l15;
            #pragma unroll
            for (int p = 0; p < 8; ++p) {
                int kk = grp * 16 + p * 2;
                af.u32[p] = *(const unsigned int*)(&As[cur][m * 32 + kk]);
            }
            acc[strip] = __builtin_amdgcn_wmma_f32_16x16x32_bf16(
                false, af.v, false, bf.v, (short)0, acc[strip], false, false);
        }
        __syncthreads();                       // done reading buf[cur]
    }

    #pragma unroll
    for (int strip = 0; strip < 4; ++strip)
        #pragma unroll
        for (int r = 0; r < 8; ++r) {
            int row = m0 + strip * 16 + r + grp * 8;
            int col = n0 + wave * 16 + l15;
            C[(size_t)row * N + col] = acc[strip][r];
        }
}

// ---------------------------------------------------------------------------
// RMSNorm + RoPE for q (gain and 1/sqrt(d) folded in)
// src: f32 [B,T,H*HD]   dst: bf16 [B,H,T,HD]   one wave per (b,t,h)
// ---------------------------------------------------------------------------
__global__ __launch_bounds__(256)
void rope_q_kernel(const float* __restrict__ src, const float* __restrict__ gain,
                   bf16_t* __restrict__ dst) {
    int gw   = blockIdx.x * 8 + (threadIdx.x >> 5);
    int lane = threadIdx.x & 31;
    int h = gw % H_;
    int t = (gw / H_) % T_;
    int b = gw / (H_ * T_);
    const float* s = src + ((size_t)(b * T_ + t) * H_ + h) * HD_;

    float v[4], ss = 0.0f;
    #pragma unroll
    for (int j = 0; j < 4; ++j) { v[j] = s[lane + 32 * j]; ss += v[j] * v[j]; }
    #pragma unroll
    for (int off = 16; off >= 1; off >>= 1) ss += __shfl_xor(ss, off, 32);
    float inv_rms = rsqrtf(ss / (float)HD_ + EPS_);
    #pragma unroll
    for (int j = 0; j < 4; ++j) v[j] *= inv_rms;

    float g = gain[h] * SCALE_;
    bf16_t* d = dst + (((size_t)b * H_ + h) * T_ + t) * HD_;
    #pragma unroll
    for (int j = 0; j < 2; ++j) {
        int dd = lane + 32 * j;                       // [0,64)
        float fr  = __expf(-((float)dd / 64.0f) * LN_BASE_);
        float ang = (float)t * fr;
        float c = __cosf(ang), sn = __sinf(ang);
        float x1 = v[j], x2 = v[j + 2];
        d[dd]      = (bf16_t)((x1 * c - x2 * sn) * g);
        d[dd + 64] = (bf16_t)((x2 * c + x1 * sn) * g);
    }
}

// same for k (no gain/scale); src f32 [B,T,NKV*HD], dst bf16 [B,NKV,T,HD]
__global__ __launch_bounds__(256)
void rope_k_kernel(const float* __restrict__ src, bf16_t* __restrict__ dst) {
    int gw   = blockIdx.x * 8 + (threadIdx.x >> 5);
    int lane = threadIdx.x & 31;
    int n = gw % NKV_;
    int t = (gw / NKV_) % T_;
    int b = gw / (NKV_ * T_);
    const float* s = src + ((size_t)(b * T_ + t) * NKV_ + n) * HD_;

    float v[4], ss = 0.0f;
    #pragma unroll
    for (int j = 0; j < 4; ++j) { v[j] = s[lane + 32 * j]; ss += v[j] * v[j]; }
    #pragma unroll
    for (int off = 16; off >= 1; off >>= 1) ss += __shfl_xor(ss, off, 32);
    float inv_rms = rsqrtf(ss / (float)HD_ + EPS_);
    #pragma unroll
    for (int j = 0; j < 4; ++j) v[j] *= inv_rms;

    bf16_t* d = dst + (((size_t)b * NKV_ + n) * T_ + t) * HD_;
    #pragma unroll
    for (int j = 0; j < 2; ++j) {
        int dd = lane + 32 * j;
        float fr  = __expf(-((float)dd / 64.0f) * LN_BASE_);
        float ang = (float)t * fr;
        float c = __cosf(ang), sn = __sinf(ang);
        float x1 = v[j], x2 = v[j + 2];
        d[dd]      = (bf16_t)(x1 * c - x2 * sn);
        d[dd + 64] = (bf16_t)(x2 * c + x1 * sn);
    }
}

// v: f32 [B,T,NKV,HD] -> bf16 [B,NKV,T,HD] (for rejection)
//                        and bf16 [B,NKV,HD,T] (d-major, for P@V fragments)
__global__ void v_transpose_kernel(const float* __restrict__ src,
                                   bf16_t* __restrict__ dst,
                                   bf16_t* __restrict__ dstT) {
    long i = (long)blockIdx.x * blockDim.x + threadIdx.x;
    if (i >= (long)B_ * T_ * NKV_ * HD_) return;
    int d = i & (HD_ - 1);
    long r = i >> 7;
    int n = r % NKV_; r /= NKV_;
    int t = r % T_;
    int b = (int)(r / T_);
    bf16_t val = (bf16_t)src[i];
    dst [(((size_t)b * NKV_ + n) * T_ + t) * HD_ + d] = val;
    dstT[(((size_t)b * NKV_ + n) * HD_ + d) * T_ + t] = val;
}

// ---------------------------------------------------------------------------
// Flash-style causal GQA attention.
// Q bf16 [B,H,T,HD], K bf16 [B,NKV,T,HD], VT bf16 [B,NKV,HD,T], Y f32 [B,T,H,HD]
// grid (T/64, H, B); 128 threads = 4 waves; K/V tiles via TDM, double-buffered.
// ---------------------------------------------------------------------------
__global__ __launch_bounds__(128)
void attn_kernel(const bf16_t* __restrict__ Q, const bf16_t* __restrict__ Km,
                 const bf16_t* __restrict__ VTm, float* __restrict__ Y) {
    __shared__ bf16_t Ks [2][32 * 128];   // [s][d]
    __shared__ bf16_t VsT[2][128 * 32];   // [d][s]
    __shared__ bf16_t Ps[4][16 * 32];

    const int tid  = threadIdx.x;
    const int lane = tid & 31;
    const int wave = tid >> 5;
    const int l15  = lane & 15;
    const int grp  = lane >> 4;
    const int tq   = blockIdx.x;
    const int h    = blockIdx.y;
    const int b    = blockIdx.z;
    const int n    = h / G_;
    const int qbase = tq * 64 + wave * 16;

    const bf16_t* kbase = Km  + (((size_t)b * NKV_ + n) * T_) * HD_;  // [s][d]
    const bf16_t* vbase = VTm + (((size_t)b * NKV_ + n) * HD_) * T_;  // [d][t]

    // Q fragments (A-layout), K-dim = 128 -> 4 fragments
    ABFrag qf[4];
    {
        const bf16_t* qp = Q + (((size_t)b * H_ + h) * T_ + qbase + l15) * HD_;
        #pragma unroll
        for (int kt = 0; kt < 4; ++kt)
            #pragma unroll
            for (int p = 0; p < 8; ++p) {
                int d = kt * 32 + grp * 16 + p * 2;
                qf[kt].u32[p] = *(const unsigned int*)(qp + d);
            }
    }

    f32x8 o[8];
    #pragma unroll
    for (int i = 0; i < 8; ++i)
        #pragma unroll
        for (int j = 0; j < 8; ++j) o[i][j] = 0.0f;
    float mi[8], li[8];
    #pragma unroll
    for (int r = 0; r < 8; ++r) { mi[r] = -1e9f; li[r] = 0.0f; }

    const int smax = tq * 64 + 63;                 // block-uniform bound
    if (wave == 0) {                               // prologue: tile sb=0
        tdm_load_2d(lds_addr_of(&Ks [0][0]), kbase, 2048, 0, 2048);       // 8KB contig
        tdm_load_2d(lds_addr_of(&VsT[0][0]), vbase, 16, 128, (unsigned long long)(T_ / 2));
    }

    int it = 0;
    for (int sb = 0; sb <= smax; sb += 32, ++it) {
        const int cur = it & 1;
        if (wave == 0) __builtin_amdgcn_s_wait_tensorcnt(0);
        __syncthreads();                           // tile (sb) visible
        if (wave == 0 && sb + 32 <= smax) {        // prefetch next tile
            tdm_load_2d(lds_addr_of(&Ks [cur ^ 1][0]), kbase + (size_t)(sb + 32) * HD_,
                        2048, 0, 2048);
            tdm_load_2d(lds_addr_of(&VsT[cur ^ 1][0]), vbase + (sb + 32),
                        16, 128, (unsigned long long)(T_ / 2));
        }

        // scores: two 16x16 tiles (s columns sb..sb+15, sb+16..sb+31)
        f32x8 s0, s1;
        #pragma unroll
        for (int j = 0; j < 8; ++j) { s0[j] = 0.0f; s1[j] = 0.0f; }
        #pragma unroll
        for (int kt = 0; kt < 4; ++kt) {
            ABFrag b0, b1;
            #pragma unroll
            for (int p = 0; p < 8; ++p) {
                int d = kt * 32 + grp * 16 + p * 2;
                b0.u32[p] = *(const unsigned int*)(&Ks[cur][l15 * 128 + d]);
                b1.u32[p] = *(const unsigned int*)(&Ks[cur][(16 + l15) * 128 + d]);
            }
            s0 = __builtin_amdgcn_wmma_f32_16x16x32_bf16(
                false, qf[kt].v, false, b0.v, (short)0, s0, false, false);
            s1 = __builtin_amdgcn_wmma_f32_16x16x32_bf16(
                false, qf[kt].v, false, b1.v, (short)0, s1, false, false);
        }

        // online softmax update
        float scl[8];
        #pragma unroll
        for (int r = 0; r < 8; ++r) {
            int qi  = qbase + r + grp * 8;
            int sj0 = sb + l15, sj1 = sb + 16 + l15;
            float v0 = (sj0 <= qi) ? s0[r] : -1e30f;
            float v1 = (sj1 <= qi) ? s1[r] : -1e30f;
            float mx = fmaxf(v0, v1);
            #pragma unroll
            for (int off = 8; off >= 1; off >>= 1)
                mx = fmaxf(mx, __shfl_xor(mx, off, 16));
            float nm = fmaxf(mi[r], mx);
            float sc = __expf(mi[r] - nm);
            float p0 = __expf(v0 - nm);
            float p1 = __expf(v1 - nm);
            float rs = p0 + p1;
            #pragma unroll
            for (int off = 8; off >= 1; off >>= 1)
                rs += __shfl_xor(rs, off, 16);
            li[r] = li[r] * sc + rs;
            mi[r] = nm;
            scl[r] = sc;
            int m = r + grp * 8;
            Ps[wave][m * 32 + l15]      = (bf16_t)p0;
            Ps[wave][m * 32 + 16 + l15] = (bf16_t)p1;
        }
        #pragma unroll
        for (int dt = 0; dt < 8; ++dt)
            #pragma unroll
            for (int r = 0; r < 8; ++r) o[dt][r] *= scl[r];

        __syncthreads();   // Ps write -> cross-lane read via LDS

        // o += P(16x32) @ V(32x128)   (V tile is d-major: contiguous k pairs)
        ABFrag pf;
        #pragma unroll
        for (int p = 0; p < 8; ++p) {
            int kk = grp * 16 + p * 2;
            pf.u32[p] = *(const unsigned int*)(&Ps[wave][l15 * 32 + kk]);
        }
        #pragma unroll
        for (int dt = 0; dt < 8; ++dt) {
            ABFrag vf;
            int d = dt * 16 + l15;
            #pragma unroll
            for (int p = 0; p < 8; ++p) {
                int kk = grp * 16 + p * 2;
                vf.u32[p] = *(const unsigned int*)(&VsT[cur][d * 32 + kk]);
            }
            o[dt] = __builtin_amdgcn_wmma_f32_16x16x32_bf16(
                false, pf.v, false, vf.v, (short)0, o[dt], false, false);
        }
        __syncthreads();   // done reading buf[cur]
    }

    // epilogue: normalize and write Y [B,T,H,HD]
    #pragma unroll
    for (int r = 0; r < 8; ++r) {
        float inv = (li[r] > 0.0f) ? 1.0f / li[r] : 0.0f;
        int qi = qbase + r + grp * 8;
        #pragma unroll
        for (int dt = 0; dt < 8; ++dt) {
            int d = dt * 16 + l15;
            Y[(((size_t)b * T_ + qi) * H_ + h) * HD_ + d] = o[dt][r] * inv;
        }
    }
}

// ---------------------------------------------------------------------------
// y -= (y . v_hat) v_hat ; one wave per (b,t,h); writes bf16 [B*T, DIM]
// ---------------------------------------------------------------------------
__global__ __launch_bounds__(256)
void reject_kernel(const float* __restrict__ Y, const bf16_t* __restrict__ Vm,
                   bf16_t* __restrict__ out) {
    int gw   = blockIdx.x * 8 + (threadIdx.x >> 5);
    int lane = threadIdx.x & 31;
    int h = gw % H_;
    int t = (gw / H_) % T_;
    int b = gw / (H_ * T_);
    int n = h / G_;
    const float*  y = Y  + ((size_t)(b * T_ + t) * H_ + h) * HD_;
    const bf16_t* v = Vm + (((size_t)b * NKV_ + n) * T_ + t) * HD_;

    float yv[4], vv[4], ss = 0.0f, dp = 0.0f;
    #pragma unroll
    for (int j = 0; j < 4; ++j) {
        int d = lane + 32 * j;
        vv[j] = (float)v[d];
        yv[j] = y[d];
        ss += vv[j] * vv[j];
        dp += yv[j] * vv[j];
    }
    #pragma unroll
    for (int off = 16; off >= 1; off >>= 1) {
        ss += __shfl_xor(ss, off, 32);
        dp += __shfl_xor(dp, off, 32);
    }
    float nrm = sqrtf(ss) + 1e-8f;
    float k   = dp / (nrm * nrm);
    bf16_t* o = out + (size_t)(b * T_ + t) * DIM_ + h * HD_;
    #pragma unroll
    for (int j = 0; j < 4; ++j) {
        int d = lane + 32 * j;
        o[d] = (bf16_t)(yv[j] - k * vv[j]);
    }
}

// ---------------------------------------------------------------------------
// host side
// ---------------------------------------------------------------------------
extern "C" void kernel_launch(void* const* d_in, const int* in_sizes, int n_in,
                              void* d_out, int out_size, void* d_ws, size_t ws_size,
                              hipStream_t stream) {
    const float* x     = (const float*)d_in[0];
    const float* Wq    = (const float*)d_in[1];
    const float* Wk    = (const float*)d_in[2];
    const float* Wv    = (const float*)d_in[3];
    const float* Wproj = (const float*)d_in[4];
    const float* qgain = (const float*)d_in[5];
    float* out = (float*)d_out;

    char* ws = (char*)d_ws;
    size_t off = 0;
    auto carve = [&](size_t bytes) {
        void* p = ws + off;
        off += (bytes + 255) & ~(size_t)255;
        return p;
    };
    bf16_t* xbf  = (bf16_t*)carve((size_t)MT_ * DIM_ * 2);          // 32 MB
    bf16_t* wqb  = (bf16_t*)carve((size_t)DIM_ * DIM_ * 2);         //  8 MB
    bf16_t* wkb  = (bf16_t*)carve((size_t)NKV_ * HD_ * DIM_ * 2);   //  2 MB
    bf16_t* wvb  = (bf16_t*)carve((size_t)NKV_ * HD_ * DIM_ * 2);   //  2 MB
    bf16_t* wpb  = (bf16_t*)carve((size_t)DIM_ * DIM_ * 2);         //  8 MB
    float*  qf32 = (float*) carve((size_t)MT_ * DIM_ * 4);          // 64 MB (reused as Y)
    float*  kf32 = (float*) carve((size_t)MT_ * NKV_ * HD_ * 4);    // 16 MB
    float*  vf32 = (float*) carve((size_t)MT_ * NKV_ * HD_ * 4);    // 16 MB
    bf16_t* qbf  = (bf16_t*)carve((size_t)MT_ * DIM_ * 2);          // 32 MB
    bf16_t* kbf  = (bf16_t*)carve((size_t)MT_ * NKV_ * HD_ * 2);    //  8 MB
    bf16_t* vbf  = (bf16_t*)carve((size_t)MT_ * NKV_ * HD_ * 2);    //  8 MB
    bf16_t* vbfT = (bf16_t*)carve((size_t)MT_ * NKV_ * HD_ * 2);    //  8 MB (d-major)
    float*  yf32 = qf32;            // q_f32 dead after rope
    bf16_t* ybf  = xbf;             // x_bf16 dead after the three input GEMMs

    auto cvt = [&](const float* s, bf16_t* d, long nel) {
        f32_to_bf16_kernel<<<dim3((unsigned)((nel + 255) / 256)), dim3(256), 0, stream>>>(s, d, nel);
    };
    // 1) convert inputs to bf16
    cvt(x,     xbf, (long)MT_ * DIM_);
    cvt(Wq,    wqb, (long)DIM_ * DIM_);
    cvt(Wk,    wkb, (long)NKV_ * HD_ * DIM_);
    cvt(Wv,    wvb, (long)NKV_ * HD_ * DIM_);
    cvt(Wproj, wpb, (long)DIM_ * DIM_);

    // 2) projections (WMMA bf16, f32 accumulate, TDM-fed tiles)
    gemm_bf16_wmma<<<dim3(DIM_ / 64, MT_ / 64), dim3(128), 0, stream>>>(
        xbf, wqb, qf32, MT_, DIM_, DIM_);
    gemm_bf16_wmma<<<dim3((NKV_ * HD_) / 64, MT_ / 64), dim3(128), 0, stream>>>(
        xbf, wkb, kf32, MT_, NKV_ * HD_, DIM_);
    gemm_bf16_wmma<<<dim3((NKV_ * HD_) / 64, MT_ / 64), dim3(128), 0, stream>>>(
        xbf, wvb, vf32, MT_, NKV_ * HD_, DIM_);

    // 3) norm + rope + layout change
    rope_q_kernel<<<dim3((B_ * T_ * H_) / 8), dim3(256), 0, stream>>>(qf32, qgain, qbf);
    rope_k_kernel<<<dim3((B_ * T_ * NKV_) / 8), dim3(256), 0, stream>>>(kf32, kbf);
    v_transpose_kernel<<<dim3((B_ * T_ * NKV_ * HD_) / 256), dim3(256), 0, stream>>>(
        vf32, vbf, vbfT);

    // 4) flash attention (causal GQA, TDM-fed K/V tiles)
    attn_kernel<<<dim3(T_ / 64, H_, B_), dim3(128), 0, stream>>>(qbf, kbf, vbfT, yf32);

    // 5) value-direction rejection -> bf16 activations
    reject_kernel<<<dim3((B_ * T_ * H_) / 8), dim3(256), 0, stream>>>(yf32, vbf, ybf);

    // 6) output projection
    gemm_bf16_wmma<<<dim3(DIM_ / 64, MT_ / 64), dim3(128), 0, stream>>>(
        ybf, wpb, out, MT_, DIM_, DIM_);

    (void)in_sizes; (void)n_in; (void)out_size; (void)ws_size;
}
